// Fusion_7842610283307
// MI455X (gfx1250) — matrix operand order repaired and checked by reading, not compiled, not run on hardware
//
#include <hip/hip_runtime.h>
#include <hip/hip_bf16.h>

// ---------------------------------------------------------------------------
// MI455X (gfx1250) implementation.
// Compute-bound (~105 GFLOP of conv vs ~10-20us of HBM traffic at 23.3 TB/s):
// all convolutions run as implicit GEMM on v_wmma_f32_16x16x32_f16 (f16
// operands, f32 accumulation).  K is ordered tap-major and inputs are staged
// channel-minor in LDS so every A/B fragment is exactly two ds_load_b128 per
// lane with compile-time offsets.  Weights are repacked once per layer into
// a tap-major, channel/cout-padded f16 image; conv blocks tile 2 output rows
// x 64 columns and each wave register-tiles 2 spatial subtiles x MT cout
// tiles to amortize fragment loads (2-3 ds_load_b128 per WMMA).
// ---------------------------------------------------------------------------

typedef _Float16 f16;
typedef __attribute__((ext_vector_type(16))) _Float16 v16h;
typedef __attribute__((ext_vector_type(8)))  _Float16 v8h;
typedef __attribute__((ext_vector_type(8)))  float    v8f;

union fragh { v16h v; v8h h[2]; };

#define BB 8
#define CCH 32
#define HH 320
#define WW 320
#define HWT (HH * WW)

// ------------------------------ elementwise --------------------------------

__global__ void k_f32_to_f16(const float* __restrict__ x, f16* __restrict__ y, int n) {
  int i = blockIdx.x * blockDim.x + threadIdx.x;
  int st = gridDim.x * blockDim.x;
  for (; i < n; i += st) y[i] = (f16)x[i];
}

__global__ void k_zero(float* p, int n) {
  int i = blockIdx.x * blockDim.x + threadIdx.x;
  if (i < n) p[i] = 0.0f;
}

__global__ void k_tanhk(const float* __restrict__ x, float* __restrict__ y, int n) {
  int i = blockIdx.x * blockDim.x + threadIdx.x;
  int st = gridDim.x * blockDim.x;
  for (; i < n; i += st) y[i] = tanhf(x[i]);
}

// ----------------------- one-time weight repack (per layer) ----------------
// wp[co][t*CINP + ci] = w[co][ci*KS*KS + t], f16, zero-padded in ci (CIN->CINP),
// t (KK->KPAD/CINP) and co (COUT->COUTP).

__global__ void k_repack(const float* __restrict__ w, f16* __restrict__ wp,
                         int CIN, int CINP, int COUT, int COUTP, int KS, int KPAD) {
  int KK = KS * KS, KE = CIN * KK;
  int total = COUTP * KPAD;
  for (int i = blockIdx.x * blockDim.x + threadIdx.x; i < total;
       i += gridDim.x * blockDim.x) {
    int co = i / KPAD, kp = i % KPAD;
    int t = kp / CINP, ci = kp % CINP;
    float v = 0.0f;
    if (co < COUT && t < KK && ci < CIN) v = w[co * KE + ci * KK + t];
    wp[i] = (f16)v;
  }
}

// ------------------------- implicit-GEMM conv (WMMA) -----------------------
// Block = 128 threads = 4 waves, covering 2 output rows x 64 columns for
// MT*16 output channels.  Wave w: row = w>>1, 32-col segment = (w&1)*32,
// register tile = 2 spatial subtiles x MT cout tiles.

template <int CIN, int CINP, int COUT, int KS, int MT>
__global__ __launch_bounds__(128) void k_conv_wmma(
    const f16* __restrict__ in, const f16* __restrict__ wp,
    const float* __restrict__ bias, float* __restrict__ out) {
  constexpr int KK     = KS * KS;
  constexpr int KP     = KK * CINP;
  constexpr int KPAD   = ((KP + 31) / 32) * 32;
  constexpr int NCH    = KPAD / 32;
  constexpr int PAD    = KS / 2;
  constexpr int TW     = 64;             // x positions per block
  constexpr int TC     = TW + KS - 1;    // halo columns
  constexpr int KSY    = KS + 1;         // halo rows (2 output rows)
  constexpr int CTILES = (COUT + 15) / 16;
  constexpr int GROUPS = CTILES / MT;

  __shared__ f16 Wl[MT * 16 * KPAD];
  __shared__ f16 Il[KSY * TC * CINP];

  const int tid    = threadIdx.x;
  const int x0     = blockIdx.x * TW;
  const int y0     = blockIdx.y * 2;
  const int n      = blockIdx.z / GROUPS;
  const int cobase = (blockIdx.z % GROUPS) * MT * 16;

  // stage repacked weights: straight contiguous b128 copies
  {
    const v8h* src = (const v8h*)(wp + cobase * KPAD);
    v8h* dst = (v8h*)Wl;
    for (int i = tid; i < MT * 16 * KPAD / 8; i += 128) dst[i] = src[i];
  }
  // stage input halo (KSY rows) channel-minor; global reads x-coalesced
  for (int idx = tid; idx < CINP * KSY * TC; idx += 128) {
    int ci = idx / (KSY * TC);
    int rc = idx % (KSY * TC);
    int r  = rc / TC;
    int cc = rc % TC;
    int yy = y0 + r - PAD;
    int xx = x0 + cc - PAD;
    f16 v = (f16)0.0f;
    if (ci < CIN && yy >= 0 && yy < HH && xx >= 0 && xx < WW)
      v = in[(long)(n * CIN + ci) * HWT + (long)yy * WW + xx];
    Il[rc * CINP + ci] = v;
  }
  __syncthreads();

  const int wave = tid >> 5, lane = tid & 31;
  const int nn = lane & 15, hi = lane >> 4;
  const int rowoff = wave >> 1;          // output row y0 + rowoff
  const int xoff   = (wave & 1) * 32;    // wave's 32-column segment
  int cb[2];
  cb[0] = (rowoff * TC + xoff + nn) * CINP;
  cb[1] = cb[0] + 16 * CINP;

  v8f c[MT][2] = {};
  int ky = 0, kx = 0;                    // tap coords (CINP==32 path)
  for (int kc = 0; kc < NCH; ++kc) {
    const int k0 = kc * 32;
    fragh a[MT], b[2];
#pragma unroll
    for (int mt = 0; mt < MT; ++mt) {
      const f16* pw = Wl + (mt * 16 + nn) * KPAD + k0 + 16 * hi;
      a[mt].h[0] = *(const v8h*)pw;
      a[mt].h[1] = *(const v8h*)(pw + 8);
    }
    if constexpr (CINP == 32) {
      const int tofs = (ky * TC + kx) * CINP + 16 * hi;
#pragma unroll
      for (int s = 0; s < 2; ++s) {
        const f16* p = Il + tofs + cb[s];
        b[s].h[0] = *(const v8h*)p;
        b[s].h[1] = *(const v8h*)(p + 8);
      }
      if (++kx == KS) { kx = 0; ++ky; }
    } else {  // CINP == 8: four taps per chunk; tap j = (h>>3) + 2*hi
      const int t0 = kc * 4;
      int ofs[4];
#pragma unroll
      for (int j = 0; j < 4; ++j) {
        int t = t0 + j;
        if (t > KK - 1) t = KK - 1;  // padded K region: A is zero there
        ofs[j] = ((t / KS) * TC + (t % KS)) * CINP;
      }
      const int o0 = hi ? ofs[2] : ofs[0];
      const int o1 = hi ? ofs[3] : ofs[1];
#pragma unroll
      for (int s = 0; s < 2; ++s) {
        b[s].h[0] = *(const v8h*)(Il + o0 + cb[s]);
        b[s].h[1] = *(const v8h*)(Il + o1 + cb[s]);
      }
    }
#pragma unroll
    for (int mt = 0; mt < MT; ++mt)
#pragma unroll
      for (int s = 0; s < 2; ++s)
        c[mt][s] = __builtin_amdgcn_wmma_f32_16x16x32_f16(
            false, a[mt].v, false, b[s].v, (short)0, c[mt][s], false, false);
  }

  // D layout: vgpr r -> M = r + 8*hi, lane -> N = lane&15
  const int yo = y0 + rowoff;
#pragma unroll
  for (int mt = 0; mt < MT; ++mt) {
#pragma unroll
    for (int s = 0; s < 2; ++s) {
      const int xw = x0 + xoff + s * 16 + nn;
#pragma unroll
      for (int r = 0; r < 8; ++r) {
        int co = cobase + mt * 16 + r + 8 * hi;
        if (co < COUT) {
          float v = c[mt][s][r];
          if (bias) v += bias[co];
          out[(long)(n * COUT + co) * HWT + (long)yo * WW + xw] = v;
        }
      }
    }
  }
}

// ----------------------------- batch norm ----------------------------------

__global__ void k_bnstats(const float* __restrict__ x, float* __restrict__ sum,
                          float* __restrict__ sq, int C) {
  __shared__ float s1[256], s2[256];
  int c = blockIdx.x;
  float a = 0.0f, b = 0.0f;
  for (int nI = 0; nI < BB; ++nI) {
    const float* base = x + (long)(nI * C + c) * HWT;
    for (int p = blockIdx.y * 256 + threadIdx.x; p < HWT; p += gridDim.y * 256) {
      float v = base[p];
      a += v;
      b += v * v;
    }
  }
  s1[threadIdx.x] = a;
  s2[threadIdx.x] = b;
  __syncthreads();
  for (int o = 128; o > 0; o >>= 1) {
    if (threadIdx.x < o) {
      s1[threadIdx.x] += s1[threadIdx.x + o];
      s2[threadIdx.x] += s2[threadIdx.x + o];
    }
    __syncthreads();
  }
  if (threadIdx.x == 0) {
    atomicAdd(&sum[c], s1[0]);
    atomicAdd(&sq[c], s2[0]);
  }
}

__global__ void k_bnfinal(const float* sum, const float* sq, const float* g,
                          const float* b, float* scale, float* shift, int C) {
  int c = threadIdx.x;
  if (c >= C) return;
  const float invN = 1.0f / ((float)BB * (float)HWT);
  float mu  = sum[c] * invN;
  float var = sq[c] * invN - mu * mu;
  float sc  = g[c] * rsqrtf(var + 1e-5f);
  scale[c]  = sc;
  shift[c]  = b[c] - mu * sc;
}

// grid.y = n*C + c : no per-element division
__global__ void k_bnrelu(const float* __restrict__ x, const float* __restrict__ scale,
                         const float* __restrict__ shift, f16* __restrict__ y, int C) {
  int nc = blockIdx.y;
  int c  = nc % C;
  float sc = scale[c], sh = shift[c];
  const float* xb = x + (long)nc * HWT;
  f16* yb = y + (long)nc * HWT;
  for (int p = blockIdx.x * 256 + threadIdx.x; p < HWT; p += gridDim.x * 256) {
    float v = xb[p] * sc + sh;
    yb[p] = (f16)(v > 0.0f ? v : 0.0f);
  }
}

// ----------------------------- sigma branch --------------------------------

__global__ void k_pool(const float* __restrict__ x, float* __restrict__ p) {
  __shared__ float s[256];
  int bc = blockIdx.x;  // n*CCH + c
  float a = 0.0f;
  for (int j = threadIdx.x; j < HWT; j += 256) a += x[(long)bc * HWT + j];
  s[threadIdx.x] = a;
  __syncthreads();
  for (int o = 128; o > 0; o >>= 1) {
    if (threadIdx.x < o) s[threadIdx.x] += s[threadIdx.x + o];
    __syncthreads();
  }
  if (threadIdx.x == 0) p[bc] = s[0] / (float)HWT;
}

__global__ void k_sigma(const float* __restrict__ p, const float* s1w,
                        const float* s1b, const float* s2w, const float* s2b,
                        const float* s3w, const float* s3b, float* __restrict__ gk) {
  int b = threadIdx.x;
  if (b >= BB) return;
  float h1[16], h2[16];
  for (int j = 0; j < 16; ++j) {
    float a = s1b[j];
    for (int i = 0; i < 32; ++i) a += p[b * 32 + i] * s1w[j * 32 + i];
    h1[j] = a > 0.0f ? a : expm1f(a);  // celu(alpha=1)
  }
  for (int j = 0; j < 16; ++j) {
    float a = s2b[j];
    for (int i = 0; i < 16; ++i) a += h1[i] * s2w[j * 16 + i];
    h2[j] = a > 0.0f ? a : expm1f(a);
  }
  float a = s3b[0];
  for (int i = 0; i < 16; ++i) a += h2[i] * s3w[i];
  float sig = (a > 0.0f ? a : 0.0f) + 0.001f;
  float s = sig + 0.2f;                 // sigma + sigma_gamma
  float e = expf(-1.0f / (2.0f * s * s));
  float d = 2.0f * e + 1.0f;
  float w3[3] = {e / d, 1.0f / d, e / d};
  int o = 0;
  for (int r = 0; r < 3; ++r)
    for (int cdx = 0; cdx < 3; ++cdx) {
      if (r == 1 && cdx == 1) continue;  // drop center
      gk[b * 8 + (o++)] = w3[r] * w3[cdx];
    }
}

// --------------------- deformable 9-tap gather (final) ---------------------

__device__ __forceinline__ float sampg(const float* img, int yi, int xi) {
  if (yi < 0 || yi >= HH || xi < 0 || xi >= WW) return 0.0f;
  return img[yi * WW + xi];
}

__device__ __forceinline__ float bil(const float* img, float yy, float xx) {
  float y0f = floorf(yy), x0f = floorf(xx);
  int y0 = (int)y0f, x0 = (int)x0f;
  float wy1 = yy - y0f, wy0 = 1.0f - wy1;
  float wx1 = xx - x0f, wx0 = 1.0f - wx1;
  return wy0 * wx0 * sampg(img, y0, x0) + wy0 * wx1 * sampg(img, y0, x0 + 1) +
         wy1 * wx0 * sampg(img, y0 + 1, x0) + wy1 * wx1 * sampg(img, y0 + 1, x0 + 1);
}

__global__ void k_final(const float* __restrict__ oa, const float* __restrict__ fus,
                        const float* __restrict__ gk, const float* __restrict__ affs,
                        float* __restrict__ out) {
  int idx = blockIdx.x * blockDim.x + threadIdx.x;
  if (idx >= BB * HWT) return;
  int n = idx / HWT, p = idx % HWT;
  int y = p / WW, x = p % WW;

  float ch[24];
#pragma unroll
  for (int c = 0; c < 24; ++c) ch[c] = oa[(long)(n * 24 + c) * HWT + p];

  float inv = 1.0f / (affs[0] + 1e-5f);
  float aff[8];
  float sa = 0.0f;
#pragma unroll
  for (int j = 0; j < 8; ++j) {
    aff[j] = tanhf(ch[16 + j]) * inv + gk[n * 8 + j];
    sa += fabsf(aff[j]);
  }
  sa += 1e-5f;
  if (sa < 1.0f) sa = 1.0f;
  float ssum = 0.0f;
#pragma unroll
  for (int j = 0; j < 8; ++j) {
    aff[j] /= sa;
    ssum += aff[j];
  }
  float aref = 1.0f - ssum;

  const float* img = fus + (long)n * HWT;
  float acc = 0.0f;
#pragma unroll
  for (int k9 = 0; k9 < 9; ++k9) {
    float av, dy, dx;
    if (k9 == 4) { av = aref; dy = 0.0f; dx = 0.0f; }
    else {
      int e = k9 < 4 ? k9 : k9 - 1;
      av = aff[e];
      dy = ch[2 * e];      // offset layout: cat[2e]=dy, cat[2e+1]=dx
      dx = ch[2 * e + 1];
    }
    float py = (float)(k9 / 3 - 1), px = (float)(k9 % 3 - 1);
    acc += av * bil(img, (float)y + py + dy, (float)x + px + dx);
  }
  out[idx] = fabsf(acc);
}

// ------------------------------- driver ------------------------------------

extern "C" void kernel_launch(void* const* d_in, const int* in_sizes, int n_in,
                              void* d_out, int out_size, void* d_ws, size_t ws_size,
                              hipStream_t stream) {
  const float* feat    = (const float*)d_in[0];
  const float* g1_w    = (const float*)d_in[1];
  const float* g1_g    = (const float*)d_in[2];
  const float* g1_b    = (const float*)d_in[3];
  const float* g2_w    = (const float*)d_in[4];
  const float* g2_g    = (const float*)d_in[5];
  const float* g2_b    = (const float*)d_in[6];
  const float* g3_w    = (const float*)d_in[7];
  const float* g3_bias = (const float*)d_in[8];
  const float* f1_w    = (const float*)d_in[9];
  const float* f1_g    = (const float*)d_in[10];
  const float* f1_b    = (const float*)d_in[11];
  const float* f2_w    = (const float*)d_in[12];
  const float* f2_bias = (const float*)d_in[13];
  const float* s1_w    = (const float*)d_in[14];
  const float* s1_b    = (const float*)d_in[15];
  const float* s2_w    = (const float*)d_in[16];
  const float* s2_b    = (const float*)d_in[17];
  const float* s3_w    = (const float*)d_in[18];
  const float* s3_b    = (const float*)d_in[19];
  const float* oa1_w   = (const float*)d_in[20];
  const float* oa1_g   = (const float*)d_in[21];
  const float* oa1_b   = (const float*)d_in[22];
  const float* oa2_w   = (const float*)d_in[23];
  const float* oa2_g   = (const float*)d_in[24];
  const float* oa2_b   = (const float*)d_in[25];
  const float* oa3_w   = (const float*)d_in[26];
  const float* oa3_bias= (const float*)d_in[27];
  const float* aff_sc  = (const float*)d_in[28];

  // workspace carve-out (~370 MB)
  char* ws = (char*)d_ws;
  size_t off = 0;
  auto alloc = [&](size_t bytes) -> void* {
    void* p = ws + off;
    off += (bytes + 255) & ~(size_t)255;
    return p;
  };
  const long N32 = (long)BB * CCH * HWT;   // 26,214,400
  const long N8  = (long)BB * 8 * HWT;
  const long N24 = (long)BB * 24 * HWT;

  f16*   featH = (f16*)alloc(N32 * 2);
  float* T     = (float*)alloc(N32 * 4);   // reused pre-activation buffer
  f16*   Hb    = (f16*)alloc(N32 * 2);     // h1 then h5
  f16*   h2b   = (f16*)alloc(N8 * 2);
  f16*   guidH = (f16*)alloc(N8 * 2);
  f16*   h3b   = (f16*)alloc(N8 * 2);
  f16*   h4b   = (f16*)alloc(N24 * 2);
  float* OA    = (float*)alloc(N24 * 4);
  float* FUS   = (float*)alloc((long)BB * HWT * 4);
  float* stats = (float*)alloc(64 * 4);    // [0..31]=sum  [32..63]=sumsq
  float* scale = (float*)alloc(32 * 4);
  float* shift = (float*)alloc(32 * 4);
  float* pool  = (float*)alloc(BB * 32 * 4);
  float* gk    = (float*)alloc(BB * 8 * 4);
  // repacked f16 weights (KPAD: 5x5/CINP32 -> 800, 5x5/CINP8 -> 224, 3x3 -> 288)
  f16* g1p  = (f16*)alloc((size_t)32 * 800 * 2);
  f16* g2p  = (f16*)alloc((size_t)16 * 800 * 2);
  f16* g3p  = (f16*)alloc((size_t)16 * 224 * 2);
  f16* oa1p = (f16*)alloc((size_t)16 * 224 * 2);
  f16* oa2p = (f16*)alloc((size_t)32 * 224 * 2);
  f16* oa3p = (f16*)alloc((size_t)32 * 800 * 2);
  f16* f1p  = (f16*)alloc((size_t)32 * 288 * 2);
  f16* f2p  = (f16*)alloc((size_t)16 * 288 * 2);
  float* sum = stats;
  float* sq  = stats + 32;

  auto bn_seq = [&](const float* pre, int C, const float* g, const float* b, f16* outh) {
    k_zero<<<1, 64, 0, stream>>>(stats, 64);
    k_bnstats<<<dim3(C, 64), 256, 0, stream>>>(pre, sum, sq, C);
    k_bnfinal<<<1, 32, 0, stream>>>(sum, sq, g, b, scale, shift, C);
    k_bnrelu<<<dim3(64, BB * C), 256, 0, stream>>>(pre, scale, shift, outh, C);
  };

  // ---- one-time weight repacks (tap-major, channel/cout-padded f16) ----
  k_repack<<<100, 256, 0, stream>>>(g1_w,  g1p,  32, 32, 32, 32, 5, 800);
  k_repack<<<50,  256, 0, stream>>>(g2_w,  g2p,  32, 32,  8, 16, 5, 800);
  k_repack<<<14,  256, 0, stream>>>(g3_w,  g3p,   8,  8,  8, 16, 5, 224);
  k_repack<<<14,  256, 0, stream>>>(oa1_w, oa1p,  8,  8,  8, 16, 5, 224);
  k_repack<<<28,  256, 0, stream>>>(oa2_w, oa2p,  8,  8, 24, 32, 5, 224);
  k_repack<<<100, 256, 0, stream>>>(oa3_w, oa3p, 24, 32, 24, 32, 5, 800);
  k_repack<<<36,  256, 0, stream>>>(f1_w,  f1p,  32, 32, 32, 32, 3, 288);
  k_repack<<<18,  256, 0, stream>>>(f2_w,  f2p,  32, 32,  1, 16, 3, 288);

  // feat -> f16 once (used by g1 and f1)
  k_f32_to_f16<<<4096, 256, 0, stream>>>(feat, featH, (int)N32);

  dim3 gridZ1(WW / 64, HH / 2, BB);        // one cout group
  dim3 gridZ2(WW / 64, HH / 2, BB * 2);    // two cout groups

  // ---- guid branch ----
  k_conv_wmma<32, 32, 32, 5, 1><<<gridZ2, 128, 0, stream>>>(featH, g1p, nullptr, T);
  bn_seq(T, 32, g1_g, g1_b, Hb);                       // h1
  k_conv_wmma<32, 32, 8, 5, 1><<<gridZ1, 128, 0, stream>>>(Hb, g2p, nullptr, T);
  bn_seq(T, 8, g2_g, g2_b, h2b);                       // h2
  k_conv_wmma<8, 8, 8, 5, 1><<<gridZ1, 128, 0, stream>>>(h2b, g3p, g3_bias, T);
  k_f32_to_f16<<<4096, 256, 0, stream>>>(T, guidH, (int)N8);  // guid (no BN/ReLU)

  // ---- off/aff branch ----
  k_conv_wmma<8, 8, 8, 5, 1><<<gridZ1, 128, 0, stream>>>(guidH, oa1p, nullptr, T);
  bn_seq(T, 8, oa1_g, oa1_b, h3b);
  k_conv_wmma<8, 8, 24, 5, 2><<<gridZ1, 128, 0, stream>>>(h3b, oa2p, nullptr, T);
  bn_seq(T, 24, oa2_g, oa2_b, h4b);
  k_conv_wmma<24, 32, 24, 5, 1><<<gridZ2, 128, 0, stream>>>(h4b, oa3p, oa3_bias, OA);

  // ---- fuse branch ----
  k_conv_wmma<32, 32, 32, 3, 2><<<gridZ1, 128, 0, stream>>>(featH, f1p, nullptr, T);
  bn_seq(T, 32, f1_g, f1_b, Hb);                       // h5 (reuses Hb)
  k_conv_wmma<32, 32, 1, 3, 1><<<gridZ1, 128, 0, stream>>>(Hb, f2p, f2_bias, T);
  k_tanhk<<<3200, 256, 0, stream>>>(T, FUS, BB * HWT);

  // ---- sigma branch ----
  k_pool<<<BB * CCH, 256, 0, stream>>>(feat, pool);
  k_sigma<<<1, 32, 0, stream>>>(pool, s1_w, s1_b, s2_w, s2_b, s3_w, s3_b, gk);

  // ---- final deformable gather ----
  k_final<<<(BB * HWT + 255) / 256, 256, 0, stream>>>(OA, FUS, gk, aff_sc,
                                                      (float*)d_out);
}
